// Attention_6751688589465
// MI455X (gfx1250) — matrix operand order repaired and checked by reading, not compiled, not run on hardware
//
#include <hip/hip_runtime.h>

// ---------------------------------------------------------------------------
// Problem constants (from reference): B=2, N=2048, M=512, DIM=1024, H=16, D=64
// ---------------------------------------------------------------------------
#define BATCH   2
#define SEQ_N   2048
#define SEQ_M   512
#define DIM     1024
#define HEADS   16
#define HDIM    64
#define LKV     (SEQ_N + SEQ_M)   // 2560
#define BN      (BATCH * SEQ_N)   // 4096
#define BM      (BATCH * SEQ_M)   // 1024

#define AS1 __attribute__((address_space(1)))
#define AS3 __attribute__((address_space(3)))

typedef __attribute__((ext_vector_type(16))) __bf16 bf16x16;
typedef __attribute__((ext_vector_type(2)))  __bf16 bf16x2;
typedef __attribute__((ext_vector_type(8)))  float  floatx8;
typedef __attribute__((ext_vector_type(4)))  int    i32x4;

#if __has_builtin(__builtin_amdgcn_global_load_async_to_lds_b128)
#define HAVE_ASYNC_LDS 1
#endif

__device__ __forceinline__ floatx8 zero8() {
  floatx8 z = {0.f, 0.f, 0.f, 0.f, 0.f, 0.f, 0.f, 0.f};
  return z;
}

// fp32 -> bf16 (round-to-nearest-even)
__device__ __forceinline__ unsigned short bf16b(float x) {
  unsigned u = __float_as_uint(x);
  u = (u + 0x7FFFu + ((u >> 16) & 1u)) >> 16;
  return (unsigned short)u;
}
__device__ __forceinline__ unsigned pack2bf(float a, float b) {
#if __has_builtin(__builtin_amdgcn_cvt_pk_bf16_f32)
  union { bf16x2 v; unsigned u; } cv;
  cv.v = __builtin_amdgcn_cvt_pk_bf16_f32(a, b);
  return cv.u;
#else
  return (unsigned)bf16b(a) | ((unsigned)bf16b(b) << 16);
#endif
}

__device__ __forceinline__ void wait_async0() {
#if __has_builtin(__builtin_amdgcn_s_wait_asynccnt)
  __builtin_amdgcn_s_wait_asynccnt(0);
#else
  asm volatile("s_wait_asynccnt 0x0" ::: "memory");
#endif
}

// ---------------------------------------------------------------------------
// WMMA fragment loaders for v_wmma_f32_16x16x32_bf16 (CDNA5 ISA 7.12.2):
//  A (16x32, MxK): lanes 0-15 row M=lane hold K 0..15 (this code's contiguous k order); lanes 16-31
//                  hold K 16..31; 2 bf16 per VGPR, 8 VGPRs.
//  B (32x16, KxN): lane (c = lane&15) is column N=c; same k order as A: lanes 0-15 hold K 0..15,
//                  lanes 16-31 hold K 16..31; 2 bf16 per VGPR.
//  C/D (16x16):    VGPR i, lane: element (M = i + 8*(lane>>4), N = lane&15).
// `base` points at element (row0, k0) of a row-major bf16 tile, leading dim
// `ld` (elements, even). The 8 dword reads per fragment are two contiguous
// 16B blocks, which the compiler fuses into ds_load_b128 pairs.
// ---------------------------------------------------------------------------
__device__ __forceinline__ bf16x16 frag_a(const unsigned short* base, int ld, int lane) {
  const int r = lane & 15, h = (lane >> 4) & 1;
  union { bf16x16 v; unsigned u[8]; } f;
  const unsigned short* row = base + r * ld;
#pragma unroll
  for (int j = 0; j < 8; ++j)
    f.u[j] = *(const unsigned*)(row + 16 * h + 2 * j);
  return f.v;
}
__device__ __forceinline__ bf16x16 frag_b(const unsigned short* base, int ld, int lane) {
  const int c = lane & 15, h = (lane >> 4) & 1;
  union { bf16x16 v; unsigned u[8]; } f;
  const unsigned short* row = base + c * ld;
#pragma unroll
  for (int j = 0; j < 8; ++j)
    f.u[j] = *(const unsigned*)(row + 16 * h + 2 * j);
  return f.v;
}

#define WMMA_BF16(a, b, c) \
  __builtin_amdgcn_wmma_f32_16x16x32_bf16(false, (a), false, (b), (short)0, (c), false, false)

// ---------------------------------------------------------------------------
// Generic GEMM: C[M,N] = A[M,K] @ W[N,K]^T (+ bias), fp32 in/out, bf16 WMMA.
// Block 256 threads = 8 waves; block tile 128x128; wave tile 32x64 (2x4 WMMA
// accumulators); K staged 64 at a time in padded LDS. M,N % 128 == 0, K % 64 == 0.
// ---------------------------------------------------------------------------
__global__ __launch_bounds__(256) void gemm_bf16_nt(
    const float* __restrict__ A, const float* __restrict__ W,
    float* __restrict__ C, const float* __restrict__ bias,
    int M, int N, int K)
{
  __shared__ unsigned short As[128][72];   // 18 KB (8-elt pad spreads banks)
  __shared__ unsigned short Bs[128][72];   // 18 KB
  const int tid  = threadIdx.x;
  const int lane = tid & 31;
  const int wv   = tid >> 5;
  const int wm   = wv >> 1;     // 0..3 : 32-row band
  const int wn   = wv & 1;      // 0..1 : 64-col band
  const int rowBase = blockIdx.y * 128;
  const int colBase = blockIdx.x * 128;

  floatx8 acc[2][4];
#pragma unroll
  for (int i = 0; i < 2; ++i)
#pragma unroll
    for (int t = 0; t < 4; ++t) acc[i][t] = zero8();

  for (int kt = 0; kt < K; kt += 64) {
    __syncthreads();
#pragma unroll
    for (int i = 0; i < 8; ++i) {
      int l   = tid + 256 * i;        // 2048 float4 slots per tile
      int row = l >> 4, c4 = l & 15;  // 16 float4 per 64-wide row
      const float* ap = A + (size_t)(rowBase + row) * K + kt + c4 * 4;
      const float* wp = W + (size_t)(colBase + row) * K + kt + c4 * 4;
      if (kt + 64 < K) {              // hint next K tile into cache
        __builtin_prefetch(ap + 64, 0, 1);
        __builtin_prefetch(wp + 64, 0, 1);
      }
      float4 fa = *(const float4*)ap;
      float4 fw = *(const float4*)wp;
      unsigned* pa = (unsigned*)&As[row][c4 * 4];
      pa[0] = pack2bf(fa.x, fa.y);
      pa[1] = pack2bf(fa.z, fa.w);
      unsigned* pb = (unsigned*)&Bs[row][c4 * 4];
      pb[0] = pack2bf(fw.x, fw.y);
      pb[1] = pack2bf(fw.z, fw.w);
    }
    __syncthreads();
#pragma unroll
    for (int s = 0; s < 2; ++s) {
      bf16x16 a0 = frag_a(&As[wm * 32][s * 32], 72, lane);
      bf16x16 a1 = frag_a(&As[wm * 32 + 16][s * 32], 72, lane);
#pragma unroll
      for (int t = 0; t < 4; ++t) {
        bf16x16 bt = frag_b(&Bs[wn * 64 + t * 16][s * 32], 72, lane);
        acc[0][t] = WMMA_BF16(a0, bt, acc[0][t]);
        acc[1][t] = WMMA_BF16(a1, bt, acc[1][t]);
      }
    }
  }

  const int c = lane & 15, hh = lane >> 4;
#pragma unroll
  for (int mi = 0; mi < 2; ++mi)
#pragma unroll
    for (int t = 0; t < 4; ++t) {
      const int col = colBase + wn * 64 + t * 16 + c;
      const float badd = bias ? bias[col] : 0.0f;
#pragma unroll
      for (int i = 0; i < 8; ++i) {
        const int row = rowBase + wm * 32 + mi * 16 + i + 8 * hh;
        C[(size_t)row * N + col] = acc[mi][t][i] + badd;
      }
    }
}

// ---------------------------------------------------------------------------
// RMSNorm + bf16 pack. One wave32 per (token, head, {q,k,v}); each lane owns 2
// of the 64 channels; sum-of-squares reduced with wave32 shuffles.
// Outputs: Qb [B,H,N,64], Kb/Vb [B,H,N+M,64] (x part at rows 0..N-1).
// ---------------------------------------------------------------------------
__global__ __launch_bounds__(256) void pack_qkv(
    const float* __restrict__ qkv, const float* __restrict__ qnw,
    const float* __restrict__ knw,
    unsigned short* __restrict__ Qb, unsigned short* __restrict__ Kb,
    unsigned short* __restrict__ Vb)
{
  const int wid  = blockIdx.x * 8 + (threadIdx.x >> 5);
  const int lane = threadIdx.x & 31;
  const int t  = wid % 3;            // 0=q 1=k 2=v (uniform per wave)
  const int h  = (wid / 3) & 15;
  const int tn = wid / 48;           // 0..BN-1
  const int b  = tn >> 11, n = tn & 2047;
  const int d0 = lane * 2;

  const float* src = qkv + (size_t)tn * (3 * DIM) + t * DIM + h * HDIM + d0;
  float x0 = src[0], x1 = src[1];
  float ss = x0 * x0 + x1 * x1;
#pragma unroll
  for (int m = 16; m >= 1; m >>= 1) ss += __shfl_xor(ss, m, 32);
  if (t < 2) {
    const float r = rsqrtf(ss * (1.0f / 64.0f) + 1e-6f);
    const float* nw = (t == 0) ? qnw : knw;
    x0 *= r * nw[d0];
    x1 *= r * nw[d0 + 1];
  }
  const unsigned pk = pack2bf(x0, x1);
  unsigned* dst;
  if (t == 0)
    dst = (unsigned*)(Qb + (((size_t)(b * HEADS + h)) * SEQ_N + n) * HDIM + d0);
  else if (t == 1)
    dst = (unsigned*)(Kb + (((size_t)(b * HEADS + h)) * LKV + n) * HDIM + d0);
  else
    dst = (unsigned*)(Vb + (((size_t)(b * HEADS + h)) * LKV + n) * HDIM + d0);
  *dst = pk;
}

__global__ __launch_bounds__(256) void pack_kvy(
    const float* __restrict__ kv, const float* __restrict__ knw,
    unsigned short* __restrict__ Kb, unsigned short* __restrict__ Vb)
{
  const int wid  = blockIdx.x * 8 + (threadIdx.x >> 5);
  const int lane = threadIdx.x & 31;
  const int t  = wid & 1;            // 0=k 1=v (uniform per wave)
  const int h  = (wid >> 1) & 15;
  const int tm = wid >> 5;           // 0..BM-1
  const int b  = tm >> 9, m = tm & 511;
  const int d0 = lane * 2;

  const float* src = kv + (size_t)tm * (2 * DIM) + t * DIM + h * HDIM + d0;
  float x0 = src[0], x1 = src[1];
  if (t == 0) {
    float ss = x0 * x0 + x1 * x1;
#pragma unroll
    for (int msk = 16; msk >= 1; msk >>= 1) ss += __shfl_xor(ss, msk, 32);
    const float r = rsqrtf(ss * (1.0f / 64.0f) + 1e-6f);
    x0 *= r * knw[d0];
    x1 *= r * knw[d0 + 1];
  }
  const unsigned pk = pack2bf(x0, x1);
  const size_t base = (((size_t)(b * HEADS + h)) * LKV + SEQ_N + m) * HDIM + d0;
  unsigned* dst = (unsigned*)((t == 0 ? Kb : Vb) + base);
  *dst = pk;
}

// ---------------------------------------------------------------------------
// Flash attention with online softmax, all matmuls via WMMA bf16.
// Grid: (N/128, H, B). Block = 256 threads = 8 waves; each wave owns 16 query
// rows and the full 64-wide head dim. Per 64-key tile:
//   - K tile staged via GLOBAL_LOAD_ASYNC_TO_LDS_B128 (ASYNCcnt path) when
//     available; V tile staged transposed (so the P@V B-operand reads are
//     contiguous dwords).
//   - S = Q K^T via WMMA; fp32 online softmax via wave32 shuffles.
//   - P bounced through per-wave LDS (accumulator->A-operand layout fixup);
//     wave-private, and same-wave LDS ops complete in order, so only a
//     compiler fence is needed (no block barrier).
//   - O += P V via WMMA.
// Output written as [B, N, H*D] fp32 so the projection GEMM consumes it directly.
// ---------------------------------------------------------------------------
__global__ __launch_bounds__(256) void flash_attn(
    const unsigned short* __restrict__ Qb,
    const unsigned short* __restrict__ Kb,
    const unsigned short* __restrict__ Vb,
    float* __restrict__ attn_out)
{
  __shared__ unsigned short Ks[64][80];      // K tile [key][d]; 160B rows (16B aligned)
  __shared__ unsigned short Vt[64][72];      // V tile transposed [d][key]
  __shared__ unsigned short Ps[8][16][72];   // per-wave P (16 q x 64 keys)

  const int tid  = threadIdx.x;
  const int lane = tid & 31;
  const int w    = tid >> 5;
  const int c    = lane & 15, hh = lane >> 4;
  const int h = blockIdx.y, b = blockIdx.z;
  const int bh = b * HEADS + h;
  const unsigned short* Qh = Qb + (size_t)bh * SEQ_N * HDIM;
  const unsigned short* Kh = Kb + (size_t)bh * LKV * HDIM;
  const unsigned short* Vh = Vb + (size_t)bh * LKV * HDIM;
  const int qrow0 = blockIdx.x * 128 + w * 16;

  bf16x16 qf[2];
#pragma unroll
  for (int s = 0; s < 2; ++s)
    qf[s] = frag_a(Qh + (size_t)qrow0 * HDIM + s * 32, HDIM, lane);

  float rmax[8], rsum[8];
  floatx8 ob[4];
#pragma unroll
  for (int i = 0; i < 8; ++i) { rmax[i] = -3.0e38f; rsum[i] = 0.0f; }
#pragma unroll
  for (int t = 0; t < 4; ++t) ob[t] = zero8();

  for (int l0 = 0; l0 < LKV; l0 += 64) {
    __syncthreads();
#if defined(HAVE_ASYNC_LDS)
    // K tile: 512 x 16B chunks, 2 per thread, DMA'd straight into LDS.
#pragma unroll
    for (int i = 0; i < 2; ++i) {
      int idx = tid + 256 * i;            // 0..511
      int row = idx >> 3, ch = idx & 7;   // 8 chunks per 64-elt row
      __builtin_amdgcn_global_load_async_to_lds_b128(
          (AS1 i32x4*)(Kh + (size_t)(l0 + row) * HDIM + ch * 8),
          (AS3 i32x4*)&Ks[row][ch * 8], 0, 0);
    }
#else
#pragma unroll
    for (int i = 0; i < 8; ++i) {
      int idx = tid + 256 * i;            // 2048 dwords
      int row = idx >> 5, cu = idx & 31;
      unsigned kd = *(const unsigned*)(Kh + (size_t)(l0 + row) * HDIM + cu * 2);
      *(unsigned*)&Ks[row][cu * 2] = kd;
    }
#endif
    // V tile, transposed (element-wise; feeds the P@V B operand)
#pragma unroll
    for (int i = 0; i < 8; ++i) {
      int idx = tid + 256 * i;
      int row = idx >> 5, cu = idx & 31;
      unsigned vd = *(const unsigned*)(Vh + (size_t)(l0 + row) * HDIM + cu * 2);
      Vt[cu * 2][row]     = (unsigned short)(vd & 0xFFFFu);
      Vt[cu * 2 + 1][row] = (unsigned short)(vd >> 16);
    }
#if defined(HAVE_ASYNC_LDS)
    wait_async0();
#endif
    __syncthreads();

    // S = Q K^T  (16 x 64), scaled by 1/sqrt(64)
    floatx8 sa[4];
#pragma unroll
    for (int t = 0; t < 4; ++t) sa[t] = zero8();
#pragma unroll
    for (int t = 0; t < 4; ++t)
#pragma unroll
      for (int s = 0; s < 2; ++s) {
        bf16x16 kf = frag_b(&Ks[t * 16][s * 32], 80, lane);
        sa[t] = WMMA_BF16(qf[s], kf, sa[t]);
      }
#pragma unroll
    for (int t = 0; t < 4; ++t)
#pragma unroll
      for (int i = 0; i < 8; ++i) sa[t][i] *= 0.125f;

    // online softmax stats (row i+8*hh lives on one 16-lane half)
    float nm[8], al[8];
#pragma unroll
    for (int i = 0; i < 8; ++i) {
      float v = fmaxf(fmaxf(sa[0][i], sa[1][i]), fmaxf(sa[2][i], sa[3][i]));
#pragma unroll
      for (int msk = 8; msk >= 1; msk >>= 1) v = fmaxf(v, __shfl_xor(v, msk, 32));
      float m2 = fmaxf(rmax[i], v);
      al[i] = __expf(rmax[i] - m2);
      nm[i] = m2;
    }
#pragma unroll
    for (int t = 0; t < 4; ++t)
#pragma unroll
      for (int i = 0; i < 8; ++i) sa[t][i] = __expf(sa[t][i] - nm[i]);
#pragma unroll
    for (int i = 0; i < 8; ++i) {
      float s4 = sa[0][i] + sa[1][i] + sa[2][i] + sa[3][i];
#pragma unroll
      for (int msk = 8; msk >= 1; msk >>= 1) s4 += __shfl_xor(s4, msk, 32);
      rsum[i] = rsum[i] * al[i] + s4;
      rmax[i] = nm[i];
    }
#pragma unroll
    for (int t = 0; t < 4; ++t)
#pragma unroll
      for (int i = 0; i < 8; ++i) ob[t][i] *= al[i];

    // P (accumulator layout) -> per-wave LDS -> A-operand layout.
    // Wave-private region + in-order same-wave LDS ops => compiler fence only.
#pragma unroll
    for (int t = 0; t < 4; ++t)
#pragma unroll
      for (int i = 0; i < 8; ++i)
        Ps[w][i + 8 * hh][t * 16 + c] = bf16b(sa[t][i]);
    asm volatile("" ::: "memory");

    // O += P V
#pragma unroll
    for (int s = 0; s < 2; ++s) {
      bf16x16 pf = frag_a(&Ps[w][0][s * 32], 72, lane);
#pragma unroll
      for (int t2 = 0; t2 < 4; ++t2) {
        bf16x16 vf = frag_b(&Vt[t2 * 16][s * 32], 72, lane);
        ob[t2] = WMMA_BF16(pf, vf, ob[t2]);
      }
    }
  }

  // normalize and write [B, N, H*D]
#pragma unroll
  for (int i = 0; i < 8; ++i) {
    const float inv = 1.0f / rsum[i];
    const size_t ro = (size_t)(b * SEQ_N + qrow0 + i + 8 * hh) * DIM + h * HDIM;
#pragma unroll
    for (int t2 = 0; t2 < 4; ++t2)
      attn_out[ro + t2 * 16 + c] = ob[t2][i] * inv;
  }
}

// ---------------------------------------------------------------------------
// Host launcher. Inputs (fp32): x, context, qkv_w, kv_y_w, proj_w, proj_b,
// q_norm_w, k_norm_w. Output: [B, N, DIM] fp32. Scratch usage: 100 MB.
// ---------------------------------------------------------------------------
extern "C" void kernel_launch(void* const* d_in, const int* in_sizes, int n_in,
                              void* d_out, int out_size, void* d_ws, size_t ws_size,
                              hipStream_t stream)
{
  (void)in_sizes; (void)n_in; (void)out_size; (void)ws_size;
  const float* x    = (const float*)d_in[0];
  const float* ctx  = (const float*)d_in[1];
  const float* qkvw = (const float*)d_in[2];
  const float* kvyw = (const float*)d_in[3];
  const float* pw   = (const float*)d_in[4];
  const float* pb   = (const float*)d_in[5];
  const float* qnw  = (const float*)d_in[6];
  const float* knw  = (const float*)d_in[7];
  float* out = (float*)d_out;

  char* ws = (char*)d_ws;
  size_t off = 0;
  auto carve = [&](size_t bytes) -> void* {
    void* p = ws + off;
    off += (bytes + 255) & ~(size_t)255;
    return p;
  };
  float*          qkv_buf  = (float*)carve((size_t)BN * (3 * DIM) * 4);     // 50.3 MB
  float*          kv_buf   = (float*)carve((size_t)BM * (2 * DIM) * 4);     //  8.4 MB
  unsigned short* Qb       = (unsigned short*)carve((size_t)BATCH * HEADS * SEQ_N * HDIM * 2);
  unsigned short* Kb       = (unsigned short*)carve((size_t)BATCH * HEADS * LKV * HDIM * 2);
  unsigned short* Vb       = (unsigned short*)carve((size_t)BATCH * HEADS * LKV * HDIM * 2);
  float*          attn_buf = (float*)carve((size_t)BN * DIM * 4);           // 16.8 MB

  // 1) qkv = x @ qkv_w^T : [4096, 3072]
  gemm_bf16_nt<<<dim3((3 * DIM) / 128, BN / 128), 256, 0, stream>>>(
      x, qkvw, qkv_buf, nullptr, BN, 3 * DIM, DIM);
  // 2) kv = context @ kv_y_w^T : [1024, 2048]
  gemm_bf16_nt<<<dim3((2 * DIM) / 128, BM / 128), 256, 0, stream>>>(
      ctx, kvyw, kv_buf, nullptr, BM, 2 * DIM, DIM);
  // 3) RMSNorm + bf16 pack into [B,H,seq,D]
  pack_qkv<<<(BN * HEADS * 3) / 8, 256, 0, stream>>>(qkv_buf, qnw, knw, Qb, Kb, Vb);
  pack_kvy<<<(BM * HEADS * 2) / 8, 256, 0, stream>>>(kv_buf, knw, Kb, Vb);
  // 4) flash attention -> [B, N, H*D] fp32
  flash_attn<<<dim3(SEQ_N / 128, HEADS, BATCH), 256, 0, stream>>>(Qb, Kb, Vb, attn_buf);
  // 5) out = attn @ proj_w^T + proj_b : [4096, 1024]
  gemm_bf16_nt<<<dim3(DIM / 128, BN / 128), 256, 0, stream>>>(
      attn_buf, pw, out, pb, BN, DIM, DIM);
}
